// AttentionBlock_8899172237879
// MI455X (gfx1250) — matrix operand order repaired and checked by reading, not compiled, not run on hardware
//
#include <hip/hip_runtime.h>
#include <math.h>

namespace {
constexpr int Bn  = 4;     // batch
constexpr int Cn  = 256;   // channels
constexpr int HWn = 4096;  // H*W
constexpr int NHn = 4;     // heads
constexpr int Dn  = 64;    // head dim
constexpr int Gn  = 8;     // groupnorm groups
constexpr int CPGn = 32;   // channels per group

// flash-kernel LDS layout (bytes), K/V chunk double buffers + P staging
constexpr int KROWPAD = 72;   // 64 + 8 halves   (bank-conflict padding)
constexpr int VROWPAD = 40;   // 32 + 8 halves
constexpr int PROWPAD = 40;   // 32 + 8 halves
constexpr int KBUF0 = 0;                         // 32*72*2 = 4608
constexpr int KBUF1 = KBUF0 + 32 * KROWPAD * 2;  // 4608
constexpr int VBUF0 = KBUF1 + 32 * KROWPAD * 2;  // 9216, 64*40*2 = 5120
constexpr int VBUF1 = VBUF0 + 64 * VROWPAD * 2;  // 14336
constexpr int PBUF  = VBUF1 + 64 * VROWPAD * 2;  // 19456, 4*16*40*2 = 5120
constexpr int SMEM_BYTES = PBUF + 4 * 16 * PROWPAD * 2;  // 24576
}

typedef __attribute__((ext_vector_type(16))) _Float16 v16h;
typedef __attribute__((ext_vector_type(8)))  _Float16 v8h;
typedef __attribute__((ext_vector_type(8)))  float    v8f;
typedef __attribute__((address_space(3)))    char     lds_char;

// ---------------- WMMA fragment helpers (wave32, 16x16x32 f16 -> f32) ------
// A (16x32, MxK): lane m=l&15 (rows), halves 0..7 -> K=kb..kb+7,
//                 halves 8..15 -> K=kb+16..kb+23, kb = (l&16)?8:0.
static __device__ __forceinline__ v16h load_a16x32(const _Float16* base, int ld,
                                                   int m0, int k0) {
  const int lane = threadIdx.x & 31;
  const int m  = m0 + (lane & 15);
  const int kb = k0 + ((lane & 16) ? 8 : 0);
  const _Float16* p = base + (size_t)m * ld + kb;
  v8h lo = *(const v8h*)(p);
  v8h hi = *(const v8h*)(p + 16);
  return __builtin_shufflevector(lo, hi, 0, 1, 2, 3, 4, 5, 6, 7,
                                 8, 9, 10, 11, 12, 13, 14, 15);
}

// B (32x16, KxN) loaded from a row-major [N][K] (i.e. transposed) source:
// lane n=l&15 (cols), halves 0..7 -> K=kb..kb+7, halves 8..15 -> K=kb+16..kb+23, kb = (l&16)?8:0.
static __device__ __forceinline__ v16h load_b32x16(const _Float16* baseT, int ld,
                                                   int n0, int k0) {
  const int lane = threadIdx.x & 31;
  const int n  = n0 + (lane & 15);
  const int kb = k0 + ((lane & 16) ? 8 : 0);
  const _Float16* p = baseT + (size_t)n * ld + kb;
  v8h lo = *(const v8h*)(p);
  v8h hi = *(const v8h*)(p + 16);
  return __builtin_shufflevector(lo, hi, 0, 1, 2, 3, 4, 5, 6, 7,
                                 8, 9, 10, 11, 12, 13, 14, 15);
}

static __device__ __forceinline__ v8f wmma16(v16h a, v16h b, v8f c) {
  return __builtin_amdgcn_wmma_f32_16x16x32_f16(false, a, false, b,
                                                (short)0, c, false, false);
}

// Async global->LDS 16B copy (CDNA5, tracked by ASYNCcnt). GVS addressing:
// mem = saddr + vaddr32, LDS dest = per-lane byte offset in vdst VGPR.
static __device__ __forceinline__ void async_ld_b128(unsigned lds_off,
                                                     unsigned voff,
                                                     const void* sbase) {
  asm volatile("global_load_async_to_lds_b128 %0, %1, %2"
               :: "v"(lds_off), "v"(voff), "s"(sbase) : "memory");
}

// ---------------- Kernel 1: GroupNorm + transpose to pixel-major f16 -------
// x[B][C][HW] f32 -> ht[B][HW][C] f16 (normalized, affine applied)
__global__ void gn_transpose_kernel(const float* __restrict__ x,
                                    const float* __restrict__ gw,
                                    const float* __restrict__ gb,
                                    _Float16* __restrict__ ht) {
  const int g = blockIdx.x, b = blockIdx.y;
  const int tid = threadIdx.x;
  const float* base = x + ((size_t)b * Cn + (size_t)g * CPGn) * HWn;
  const int N = CPGn * HWn;

  float s = 0.f, ss = 0.f;
  for (int i = tid; i < N; i += 256) {
    float v = base[i];
    s += v; ss += v * v;
  }
#pragma unroll
  for (int off = 16; off >= 1; off >>= 1) {
    s  += __shfl_xor(s, off, 32);
    ss += __shfl_xor(ss, off, 32);
  }
  __shared__ float red[2][8];
  __shared__ float stats[2];
  const int wave = tid >> 5, lane = tid & 31;
  if (lane == 0) { red[0][wave] = s; red[1][wave] = ss; }
  __syncthreads();
  if (tid == 0) {
    float S = 0.f, SS = 0.f;
    for (int w = 0; w < 8; ++w) { S += red[0][w]; SS += red[1][w]; }
    float mean = S / (float)N;
    float var  = SS / (float)N - mean * mean;
    stats[0] = mean;
    stats[1] = rsqrtf(var + 1e-5f);
  }
  __syncthreads();
  const float mean = stats[0], inv = stats[1];

  __shared__ float tile[CPGn][65];  // pad to dodge bank conflicts
  const int tp = tid & 63, tc = tid >> 6;    // coalesced global read indices
  const int tcw = tid & 31, tpw = tid >> 5;  // coalesced f16 write indices
  const float gwv = gw[g * CPGn + tcw];
  const float gbv = gb[g * CPGn + tcw];
  for (int p0 = 0; p0 < HWn; p0 += 64) {
    for (int cc = tc; cc < CPGn; cc += 4)
      tile[cc][tp] = base[(size_t)cc * HWn + p0 + tp];
    __syncthreads();
    for (int pp = tpw; pp < 64; pp += 8) {
      float v = (tile[tcw][pp] - mean) * inv * gwv + gbv;
      ht[((size_t)b * HWn + p0 + pp) * Cn + g * CPGn + tcw] = (_Float16)v;
    }
    __syncthreads();
  }
}

// ---------------- Kernel 2: weight f32 -> f16 ------------------------------
__global__ void cvt_weights_kernel(const float* __restrict__ qkvw,
                                   const float* __restrict__ projw,
                                   _Float16* __restrict__ wq,
                                   _Float16* __restrict__ wp) {
  int i = blockIdx.x * blockDim.x + threadIdx.x;
  if (i < 3 * Cn * Cn) wq[i] = (_Float16)qkvw[i];
  if (i < Cn * Cn)     wp[i] = (_Float16)projw[i];
}

// ---------------- Kernel 3: qkv GEMM (WMMA) --------------------------------
// qkvT[p][o] = sum_c ht[p][c] * wq[o][c] + bias[o], scattered to
// Qt[B][NH][HW][D], Kt[B][NH][HW][D], V[B][NH][D][HW] (all f16)
__global__ void qkv_gemm_kernel(const _Float16* __restrict__ ht,
                                const _Float16* __restrict__ wq,
                                const float* __restrict__ qkvb,
                                _Float16* __restrict__ qt,
                                _Float16* __restrict__ kt,
                                _Float16* __restrict__ vv) {
  const int nblk = blockIdx.x;   // 0..11 : 64-wide output-channel block
  const int pblk = blockIdx.y;   // 0..63 : 64-wide pixel block
  const int b    = blockIdx.z;
  const int wave = threadIdx.x >> 5, lane = threadIdx.x & 31;
  const int m0 = pblk * 64 + wave * 16;
  const int n0 = nblk * 64;
  const _Float16* A = ht + (size_t)b * HWn * Cn;

  v8f acc[4] = {};
#pragma unroll
  for (int k0 = 0; k0 < Cn; k0 += 32) {
    v16h a = load_a16x32(A, Cn, m0, k0);
    v16h b0 = load_b32x16(wq, Cn, n0 +  0, k0);
    v16h b1 = load_b32x16(wq, Cn, n0 + 16, k0);
    v16h b2 = load_b32x16(wq, Cn, n0 + 32, k0);
    v16h b3 = load_b32x16(wq, Cn, n0 + 48, k0);
    acc[0] = wmma16(a, b0, acc[0]);
    acc[1] = wmma16(a, b1, acc[1]);
    acc[2] = wmma16(a, b2, acc[2]);
    acc[3] = wmma16(a, b3, acc[3]);
  }

  const int hi = (lane & 16) ? 8 : 0, n = lane & 15;
  const int region = nblk >> 2;      // 0=Q 1=K 2=V (64-aligned => uniform)
  const int head   = nblk & 3;
#pragma unroll
  for (int t = 0; t < 4; ++t) {
#pragma unroll
    for (int r = 0; r < 8; ++r) {
      const int o = n0 + t * 16 + n;
      const int p = m0 + r + hi;
      const int dcol = t * 16 + n;   // o % 64
      _Float16 hv = (_Float16)(acc[t][r] + qkvb[o]);
      if (region == 2) {
        vv[(((size_t)b * NHn + head) * Dn + dcol) * HWn + p] = hv;
      } else {
        _Float16* dst = (region == 0) ? qt : kt;
        dst[(((size_t)b * NHn + head) * HWn + p) * Dn + dcol] = hv;
      }
    }
  }
}

// ---------------- Kernel 4: flash attention --------------------------------
// Per wave: 16-row strip of Q^T. K/V chunks (32 keys) are double-buffered in
// LDS via async global->LDS DMA shared by all 4 waves of the block.
__global__ void flash_attn_kernel(const _Float16* __restrict__ qt,
                                  const _Float16* __restrict__ kt,
                                  const _Float16* __restrict__ vv,
                                  _Float16* __restrict__ oT) {
  __shared__ __align__(16) char smem[SMEM_BYTES];
  const unsigned smem_lds = (unsigned)(size_t)(lds_char*)smem;

  const int sblk = blockIdx.x;   // 64 strip-blocks of 64 rows
  const int h = blockIdx.y, b = blockIdx.z;
  const int tid = threadIdx.x;
  const int wave = tid >> 5, lane = tid & 31;
  const int i0 = sblk * 64 + wave * 16;
  const _Float16* qth = qt + ((size_t)b * NHn + h) * HWn * Dn;
  const _Float16* kth = kt + ((size_t)b * NHn + h) * HWn * Dn;
  const _Float16* vh  = vv + ((size_t)b * NHn + h) * Dn * HWn;

  // Per-thread async-copy assignments (4 x b128 per thread per chunk):
  // K chunk: 32 rows x 64 halves -> rows padded to KROWPAD in LDS
  const int krow0 = tid >> 3,         kseg0 = tid & 7;         // l = tid
  const int krow1 = (tid + 128) >> 3, kseg1 = tid & 7;         // l = tid+128
  // V chunk: 64 rows x 32 halves -> rows padded to VROWPAD in LDS
  const int vrow0 = tid >> 2,         vseg0 = tid & 3;
  const int vrow1 = (tid + 128) >> 2, vseg1 = tid & 3;

  auto issue_chunk = [&](int j0, int bufsel) {
    const unsigned kb = smem_lds + (bufsel ? KBUF1 : KBUF0);
    const unsigned vb = smem_lds + (bufsel ? VBUF1 : VBUF0);
    async_ld_b128(kb + (unsigned)(krow0 * KROWPAD + kseg0 * 8) * 2,
                  (unsigned)(((j0 + krow0) * Dn + kseg0 * 8) * 2), kth);
    async_ld_b128(kb + (unsigned)(krow1 * KROWPAD + kseg1 * 8) * 2,
                  (unsigned)(((j0 + krow1) * Dn + kseg1 * 8) * 2), kth);
    async_ld_b128(vb + (unsigned)(vrow0 * VROWPAD + vseg0 * 8) * 2,
                  (unsigned)((vrow0 * HWn + j0 + vseg0 * 8) * 2), vh);
    async_ld_b128(vb + (unsigned)(vrow1 * VROWPAD + vseg1 * 8) * 2,
                  (unsigned)((vrow1 * HWn + j0 + vseg1 * 8) * 2), vh);
  };

  const v16h aq0 = load_a16x32(qth, Dn, i0, 0);
  const v16h aq1 = load_a16x32(qth, Dn, i0, 32);

  float mrow[8], lrow[8];
#pragma unroll
  for (int r = 0; r < 8; ++r) { mrow[r] = -1e30f; lrow[r] = 0.f; }
  v8f oacc[4] = {};
  const int rowb = (lane & 16) ? 8 : 0, n = lane & 15;
  const float scale = 0.125f;  // D^-0.5
  _Float16* pw = (_Float16*)(smem + PBUF) + wave * 16 * PROWPAD;

  issue_chunk(0, 0);
  constexpr int NITER = HWn / 32;
  for (int it = 0; it < NITER; ++it) {
    const int cur = it & 1;
    if (it + 1 < NITER) {
      issue_chunk((it + 1) * 32, cur ^ 1);
      // 4 outstanding belong to the next buffer; in-order completion
      // guarantees the current buffer's DMA is done.
      asm volatile("s_wait_asynccnt 0x4" ::: "memory");
    } else {
      asm volatile("s_wait_asynccnt 0x0" ::: "memory");
    }
    __syncthreads();  // current chunk visible to all waves

    const _Float16* kbuf = (const _Float16*)(smem + (cur ? KBUF1 : KBUF0));
    const _Float16* vbuf = (const _Float16*)(smem + (cur ? VBUF1 : VBUF0));

    // S tiles: 16 rows x 32 keys; K-loop over d=64 (two WMMAs per tile)
    v16h bk00 = load_b32x16(kbuf, KROWPAD, 0, 0);
    v16h bk01 = load_b32x16(kbuf, KROWPAD, 0, 32);
    v16h bk10 = load_b32x16(kbuf, KROWPAD, 16, 0);
    v16h bk11 = load_b32x16(kbuf, KROWPAD, 16, 32);
    v8f s1 = {}, s2 = {};
    s1 = wmma16(aq0, bk00, s1);
    s1 = wmma16(aq1, bk01, s1);
    s2 = wmma16(aq0, bk10, s2);
    s2 = wmma16(aq1, bk11, s2);

    // Issue V fragments early; DS latency hides under the softmax VALU work.
    v16h bv0 = load_b32x16(vbuf, VROWPAD,  0, 0);
    v16h bv1 = load_b32x16(vbuf, VROWPAD, 16, 0);
    v16h bv2 = load_b32x16(vbuf, VROWPAD, 32, 0);
    v16h bv3 = load_b32x16(vbuf, VROWPAD, 48, 0);

    float alpha[8];
#pragma unroll
    for (int r = 0; r < 8; ++r) {
      s1[r] *= scale; s2[r] *= scale;
      float mx = fmaxf(s1[r], s2[r]);
#pragma unroll
      for (int off = 8; off >= 1; off >>= 1)
        mx = fmaxf(mx, __shfl_xor(mx, off, 16));
      float mnew = fmaxf(mrow[r], mx);
      alpha[r] = __expf(mrow[r] - mnew);
      float p1 = __expf(s1[r] - mnew);
      float p2 = __expf(s2[r] - mnew);
      s1[r] = p1; s2[r] = p2;
      float rs = p1 + p2;
#pragma unroll
      for (int off = 8; off >= 1; off >>= 1)
        rs += __shfl_xor(rs, off, 16);
      lrow[r] = lrow[r] * alpha[r] + rs;
      mrow[r] = mnew;
    }
#pragma unroll
    for (int t = 0; t < 4; ++t)
#pragma unroll
      for (int r = 0; r < 8; ++r)
        oacc[t][r] *= alpha[r];

    // C-layout P (f32) -> per-wave LDS tile [16][32] f16 -> A fragment
#pragma unroll
    for (int r = 0; r < 8; ++r) {
      pw[(r + rowb) * PROWPAD + n]      = (_Float16)s1[r];
      pw[(r + rowb) * PROWPAD + 16 + n] = (_Float16)s2[r];
    }
    asm volatile("s_wait_dscnt 0" ::: "memory");
    v16h ap = load_a16x32(pw, PROWPAD, 0, 0);

    // O(16x64) += P(16x32) x V^T(32x64)
    oacc[0] = wmma16(ap, bv0, oacc[0]);
    oacc[1] = wmma16(ap, bv1, oacc[1]);
    oacc[2] = wmma16(ap, bv2, oacc[2]);
    oacc[3] = wmma16(ap, bv3, oacc[3]);

    __syncthreads();  // everyone done with `cur` before it is refilled
  }

  float invl[8];
#pragma unroll
  for (int r = 0; r < 8; ++r) invl[r] = 1.f / lrow[r];
  _Float16* od = oT + (size_t)b * HWn * Cn + (size_t)h * Dn;
#pragma unroll
  for (int t = 0; t < 4; ++t)
#pragma unroll
    for (int r = 0; r < 8; ++r)
      od[(size_t)(i0 + r + rowb) * Cn + t * 16 + n] =
          (_Float16)(oacc[t][r] * invl[r]);
}

// ---------------- Kernel 5: proj GEMM + bias + residual (WMMA) -------------
__global__ void proj_residual_kernel(const _Float16* __restrict__ oT,
                                     const _Float16* __restrict__ wp,
                                     const float* __restrict__ pb,
                                     const float* __restrict__ x,
                                     float* __restrict__ out) {
  const int nblk = blockIdx.x;   // 0..3
  const int pblk = blockIdx.y;   // 0..63
  const int b    = blockIdx.z;
  const int wave = threadIdx.x >> 5, lane = threadIdx.x & 31;
  const int m0 = pblk * 64 + wave * 16;
  const int n0 = nblk * 64;
  const _Float16* A = oT + (size_t)b * HWn * Cn;

  v8f acc[4] = {};
#pragma unroll
  for (int k0 = 0; k0 < Cn; k0 += 32) {
    v16h a = load_a16x32(A, Cn, m0, k0);
    v16h b0 = load_b32x16(wp, Cn, n0 +  0, k0);
    v16h b1 = load_b32x16(wp, Cn, n0 + 16, k0);
    v16h b2 = load_b32x16(wp, Cn, n0 + 32, k0);
    v16h b3 = load_b32x16(wp, Cn, n0 + 48, k0);
    acc[0] = wmma16(a, b0, acc[0]);
    acc[1] = wmma16(a, b1, acc[1]);
    acc[2] = wmma16(a, b2, acc[2]);
    acc[3] = wmma16(a, b3, acc[3]);
  }

  const int hi = (lane & 16) ? 8 : 0, n = lane & 15;
#pragma unroll
  for (int t = 0; t < 4; ++t)
#pragma unroll
    for (int r = 0; r < 8; ++r) {
      const int o = n0 + t * 16 + n;
      const int p = m0 + r + hi;
      const size_t idx = ((size_t)b * Cn + o) * HWn + p;
      out[idx] = x[idx] + acc[t][r] + pb[o];
    }
}

// ---------------- Host launcher --------------------------------------------
extern "C" void kernel_launch(void* const* d_in, const int* in_sizes, int n_in,
                              void* d_out, int out_size, void* d_ws, size_t ws_size,
                              hipStream_t stream) {
  (void)in_sizes; (void)n_in; (void)out_size; (void)ws_size;
  const float* x      = (const float*)d_in[0];
  const float* norm_w = (const float*)d_in[1];
  const float* norm_b = (const float*)d_in[2];
  const float* qkv_w  = (const float*)d_in[3];
  const float* qkv_b  = (const float*)d_in[4];
  const float* proj_w = (const float*)d_in[5];
  const float* proj_b = (const float*)d_in[6];
  float* out = (float*)d_out;

  char* ws = (char*)d_ws;
  size_t off = 0;
  auto wsalloc = [&](size_t bytes) -> void* {
    void* p = ws + off;
    off = (off + bytes + 255) & ~(size_t)255;
    return p;
  };
  _Float16* ht   = (_Float16*)wsalloc((size_t)Bn * HWn * Cn * 2);        // 8 MB
  _Float16* qt   = (_Float16*)wsalloc((size_t)Bn * NHn * HWn * Dn * 2);  // 8 MB
  _Float16* kt   = (_Float16*)wsalloc((size_t)Bn * NHn * HWn * Dn * 2);  // 8 MB
  _Float16* vv   = (_Float16*)wsalloc((size_t)Bn * NHn * Dn * HWn * 2);  // 8 MB
  _Float16* oT   = (_Float16*)wsalloc((size_t)Bn * HWn * Cn * 2);        // 8 MB
  _Float16* wq16 = (_Float16*)wsalloc((size_t)3 * Cn * Cn * 2);
  _Float16* wp16 = (_Float16*)wsalloc((size_t)Cn * Cn * 2);

  gn_transpose_kernel<<<dim3(Gn, Bn), 256, 0, stream>>>(x, norm_w, norm_b, ht);
  cvt_weights_kernel<<<(3 * Cn * Cn + 255) / 256, 256, 0, stream>>>(
      qkv_w, proj_w, wq16, wp16);
  qkv_gemm_kernel<<<dim3(12, HWn / 64, Bn), 128, 0, stream>>>(
      ht, wq16, qkv_b, qt, kt, vv);
  flash_attn_kernel<<<dim3(HWn / 64, NHn, Bn), 128, 0, stream>>>(
      qt, kt, vv, oT);
  proj_residual_kernel<<<dim3(Cn / 64, HWn / 64, Bn), 128, 0, stream>>>(
      oT, wp16, proj_b, x, out);
}
